// SamDetector_9620726743860
// MI455X (gfx1250) — matrix operand order repaired and checked by reading, compiled, not run on hardware
//
#include <hip/hip_runtime.h>
#include <hip/hip_bf16.h>

// ---------------- constants ----------------
constexpr int NB  = 16;    // batch
constexpr int NQ  = 300;   // queries
constexpr int NQP = 304;   // queries padded to 16
constexpr int NQT = NQP / 16;
constexpr int NC  = 256;   // channels
constexpr int NHH = 64;    // H
constexpr int NWW = 64;    // W
constexpr int NHW = 4096;  // H*W
constexpr int NNH = 8;     // heads
constexpr int NHD = 32;    // head dim
constexpr int NDFF = 2048;

typedef __bf16 bf16_t;
typedef __attribute__((ext_vector_type(16))) __bf16 v16bf;
typedef __attribute__((ext_vector_type(8)))  float  v8f;

// ---------------- WMMA helpers ----------------
__device__ inline v8f wmma_bf16(v16bf a, v16bf b, v8f c) {
  // (neg_a, A, neg_b, B, c_mod, C, reuse_a, reuse_b)
  return __builtin_amdgcn_wmma_f32_16x16x32_bf16(false, a, false, b, (short)0, c, false, false);
}

// A fragment: 16x32 bf16, row-major source, elem(m,k) = A[m*lda + k]
__device__ inline v16bf load_a_rm_bf16(const bf16_t* A, int lda) {
  int l = threadIdx.x & 31, m = l & 15, hf = l >> 4;
  v16bf r;
#pragma unroll
  for (int i = 0; i < 8; ++i) {
    int k = hf * 16 + 2 * i;
    r[2 * i]     = A[(size_t)m * lda + k];
    r[2 * i + 1] = A[(size_t)m * lda + k + 1];
  }
  return r;
}
// B fragment: 32x16 bf16, row-major source, elem(k,n) = B[k*ldb + n]
__device__ inline v16bf load_b_rm_bf16(const bf16_t* Bm, int ldb) {
  int l = threadIdx.x & 31, n = l & 15, kh = l >> 4;
  v16bf r;
#pragma unroll
  for (int i = 0; i < 8; ++i) {
    int k = kh * 16 + 2 * i;
    r[2 * i]     = Bm[(size_t)k * ldb + n];
    r[2 * i + 1] = Bm[(size_t)(k + 1) * ldb + n];
  }
  return r;
}
// B fragment: col-major source, elem(k,n) = B[n*ldb + k]
__device__ inline v16bf load_b_cm_bf16(const bf16_t* Bm, int ldb) {
  int l = threadIdx.x & 31, n = l & 15, kh = l >> 4;
  const bf16_t* p = Bm + (size_t)n * ldb + kh * 16;
  v16bf r;
#pragma unroll
  for (int i = 0; i < 8; ++i) { r[2 * i] = p[2 * i]; r[2 * i + 1] = p[2 * i + 1]; }
  return r;
}
// B fragment from fp32 source (convert on the fly), row-major
__device__ inline v16bf load_b_rm_f32(const float* Bm, int ldb) {
  int l = threadIdx.x & 31, n = l & 15, kh = l >> 4;
  v16bf r;
#pragma unroll
  for (int i = 0; i < 8; ++i) {
    int k = kh * 16 + 2 * i;
    r[2 * i]     = (bf16_t)Bm[(size_t)k * ldb + n];
    r[2 * i + 1] = (bf16_t)Bm[(size_t)(k + 1) * ldb + n];
  }
  return r;
}

__device__ inline float half_max16(float v) {
#pragma unroll
  for (int off = 1; off < 16; off <<= 1) v = fmaxf(v, __shfl_xor(v, off, 32));
  return v;
}
__device__ inline float half_sum16(float v) {
#pragma unroll
  for (int off = 1; off < 16; off <<= 1) v += __shfl_xor(v, off, 32);
  return v;
}

__device__ inline float block_sum256(float v, float* red) {
  int t = threadIdx.x;
  red[t] = v; __syncthreads();
  for (int s = 128; s > 0; s >>= 1) { if (t < s) red[t] += red[t + s]; __syncthreads(); }
  float r = red[0]; __syncthreads();
  return r;
}

// ---------------- utility ----------------
__global__ void k_f32_to_bf16(const float* s, bf16_t* d, int n) {
  int t = blockIdx.x * blockDim.x + threadIdx.x;
  if (t < n) d[t] = (bf16_t)s[t];
}

// ---------------- stage A: batch-independent self-attention ----------------
// qkv projections of broadcast point embedding: sqkv[which][q][c]
__global__ void k_self_qkv(const float* pe, const float* wi, const float* bi, float* sqkv) {
  int t = blockIdx.x * blockDim.x + threadIdx.x;
  if (t >= 3 * NQ * NC) return;
  int c = t % NC, q = (t / NC) % NQ, which = t / (NQ * NC);
  int row = which * NC + c;
  const float* w = wi + (size_t)row * NC;
  const float* x = pe + (size_t)q * NC;
  float s = bi[row];
  for (int k = 0; k < NC; ++k) s += x[k] * w[k];
  sqkv[t] = s;
}

// one block per (q, head); block = 64 threads
__global__ void k_self_attn(const float* sqkv, float* attn1) {
  int q = blockIdx.x / NNH, h = blockIdx.x % NNH;
  const float* sq = sqkv;
  const float* sk = sqkv + (size_t)NQ * NC;
  const float* sv = sqkv + (size_t)2 * NQ * NC;
  __shared__ float sc[NQ];
  __shared__ float red[64];
  int t = threadIdx.x;
  const float inv = 0.17677669529663687f; // 1/sqrt(32)
  for (int j = t; j < NQ; j += 64) {
    const float* a = sq + (size_t)q * NC + h * NHD;
    const float* b = sk + (size_t)j * NC + h * NHD;
    float s = 0.f;
    for (int d = 0; d < NHD; ++d) s += a[d] * b[d];
    sc[j] = s * inv;
  }
  __syncthreads();
  float lm = -3.0e38f;
  for (int j = t; j < NQ; j += 64) lm = fmaxf(lm, sc[j]);
  red[t] = lm; __syncthreads();
  for (int s = 32; s > 0; s >>= 1) { if (t < s) red[t] = fmaxf(red[t], red[t + s]); __syncthreads(); }
  float mx = red[0]; __syncthreads();
  float ls = 0.f;
  for (int j = t; j < NQ; j += 64) { float e = expf(sc[j] - mx); sc[j] = e; ls += e; }
  red[t] = ls; __syncthreads();
  for (int s = 32; s > 0; s >>= 1) { if (t < s) red[t] += red[t + s]; __syncthreads(); }
  float inv_sum = 1.0f / red[0];
  __syncthreads();
  if (t < NHD) {
    float acc = 0.f;
    const float* vb = sv + h * NHD + t;
    for (int j = 0; j < NQ; ++j) acc += sc[j] * vb[(size_t)j * NC];
    attn1[(size_t)q * NC + h * NHD + t] = acc * inv_sum;
  }
}

// out-proj + residual + LN(n2)
__global__ void k_self_out_ln(const float* attn1, const float* wo, const float* bo,
                              const float* pe, const float* g, const float* bln, float* qbase) {
  int q = blockIdx.x, t = threadIdx.x;
  __shared__ float sh[NC]; __shared__ float red[NC];
  sh[t] = attn1[(size_t)q * NC + t]; __syncthreads();
  const float* w = wo + (size_t)t * NC;
  float v = bo[t];
  for (int k = 0; k < NC; ++k) v += sh[k] * w[k];
  v += pe[(size_t)q * NC + t];
  float mu  = block_sum256(v, red) * (1.0f / NC);
  float d   = v - mu;
  float var = block_sum256(d * d, red) * (1.0f / NC);
  qbase[(size_t)q * NC + t] = d * rsqrtf(var + 1e-5f) * g[t] + bln[t];
}

// cross-attn q projection (pre-scaled by 1/sqrt(HD)), padded to NQP rows, bf16
__global__ void k_q_proj_ca(const float* qbase, const float* wi, const float* bi, bf16_t* qpca) {
  int q = blockIdx.x, t = threadIdx.x;
  if (q >= NQ) { qpca[(size_t)q * NC + t] = (bf16_t)0.0f; return; }
  __shared__ float sh[NC];
  sh[t] = qbase[(size_t)q * NC + t]; __syncthreads();
  const float* w = wi + (size_t)t * NC;
  float v = bi[t];
  for (int k = 0; k < NC; ++k) v += sh[k] * w[k];
  qpca[(size_t)q * NC + t] = (bf16_t)(v * 0.17677669529663687f);
}

// ---------------- stage B: per-batch heavy GEMMs (WMMA) ----------------
// K^T / V^T projections: dst[b][c_out][n] = sum_c W[c_out,c]*img[b,c,n] + bias
__global__ void k_kv_proj(const float* img, const bf16_t* cawib, const float* ca_bi,
                          bf16_t* kpT, bf16_t* vpT) {
  int w = blockIdx.x * 4 + (threadIdx.x >> 5);
  int nt = w & 255, mt = (w >> 8) & 15, b = (w >> 12) & 15, which = w >> 16;
  const bf16_t* A  = cawib + (size_t)(NC + which * NC + mt * 16) * NC;
  const float*  Bp = img + (size_t)b * NC * NHW + nt * 16;
  v8f acc = {};
  for (int kk = 0; kk < NC / 32; ++kk) {
    v16bf a  = load_a_rm_bf16(A + kk * 32, NC);
    v16bf bb = load_b_rm_f32(Bp + (size_t)(kk * 32) * NHW, NHW);
    acc = wmma_bf16(a, bb, acc);
  }
  int l = threadIdx.x & 31, n = l & 15, mh = l >> 4;
  const float* bias = ca_bi + NC + which * NC + mt * 16;
  bf16_t* dst = (which ? vpT : kpT) + (size_t)b * NC * NHW + (size_t)(mt * 16) * NHW + nt * 16;
#pragma unroll
  for (int r = 0; r < 8; ++r) {
    int m = r + mh * 8;
    dst[(size_t)m * NHW + n] = (bf16_t)(acc[r] + bias[m]);
  }
}

// cross attention: one wave per (b, h, qtile). two-pass softmax, WMMA P@V.
__global__ void k_cross_attn(const bf16_t* qpca, const bf16_t* kpT, const bf16_t* vpT,
                             float* caout) {
  int qt = blockIdx.x % NQT;
  int h  = (blockIdx.x / NQT) % NNH;
  int b  = blockIdx.x / (NQT * NNH);
  __shared__ bf16_t pbuf[16 * 32];

  v16bf aq = load_a_rm_bf16(qpca + (size_t)(qt * 16) * NC + h * NHD, NC);
  const bf16_t* kb  = kpT + ((size_t)b * NC + h * NHD) * NHW;
  const bf16_t* vb0 = vpT + ((size_t)b * NC + h * NHD) * NHW;
  const bf16_t* vb1 = vpT + ((size_t)b * NC + h * NHD + 16) * NHW;
  v8f zc = {};

  float rmax[8], rsum[8];
#pragma unroll
  for (int r = 0; r < 8; ++r) { rmax[r] = -3.0e38f; rsum[r] = 0.f; }

  // pass 1: online row max / sum over 4096 keys
  for (int nt = 0; nt < NHW / 16; ++nt) {
    v8f s = wmma_bf16(aq, load_b_rm_bf16(kb + nt * 16, NHW), zc);
#pragma unroll
    for (int r = 0; r < 8; ++r) {
      float tm = half_max16(s[r]);
      float nm = fmaxf(rmax[r], tm);
      float te = half_sum16(expf(s[r] - nm));
      rsum[r] = rsum[r] * expf(rmax[r] - nm) + te;
      rmax[r] = nm;
    }
  }
  float rinv[8];
#pragma unroll
  for (int r = 0; r < 8; ++r) rinv[r] = 1.0f / rsum[r];

  int l = threadIdx.x & 31, col = l & 15, mh = l >> 4;
  v8f o0 = {}, o1 = {};
  // pass 2: recompute scores, normalize, WMMA-accumulate P@V
  for (int np = 0; np < NHW / 32; ++np) {
#pragma unroll
    for (int j = 0; j < 2; ++j) {
      v8f s = wmma_bf16(aq, load_b_rm_bf16(kb + (np * 2 + j) * 16, NHW), zc);
#pragma unroll
      for (int r = 0; r < 8; ++r) {
        float p = expf(s[r] - rmax[r]) * rinv[r];
        pbuf[(r + mh * 8) * 32 + j * 16 + col] = (bf16_t)p;
      }
    }
    __syncthreads();
    asm volatile("s_wait_dscnt 0" ::: "memory");
    v16bf ap = load_a_rm_bf16(pbuf, 32);
    o0 = wmma_bf16(ap, load_b_cm_bf16(vb0 + np * 32, NHW), o0);
    o1 = wmma_bf16(ap, load_b_cm_bf16(vb1 + np * 32, NHW), o1);
    __syncthreads();
  }
  float* dst = caout + ((size_t)b * NQP + qt * 16) * NC + h * NHD;
#pragma unroll
  for (int r = 0; r < 8; ++r) {
    int m = r + mh * 8;
    dst[(size_t)m * NC + col]      = o0[r];
    dst[(size_t)m * NC + 16 + col] = o1[r];
  }
}

// cross-attn out-proj + residual + LN(n1); emits fp32 x and bf16 xb (padded rows zero)
__global__ void k_oproj_ln(const float* caout, const float* wo, const float* bo,
                           const float* qbase, const float* g, const float* bln,
                           float* x, bf16_t* xb) {
  int bq = blockIdx.x, b = bq / NQP, q = bq % NQP, t = threadIdx.x;
  (void)b;
  size_t idx = (size_t)bq * NC + t;
  if (q >= NQ) { x[idx] = 0.f; xb[idx] = (bf16_t)0.f; return; }
  __shared__ float sh[NC]; __shared__ float red[NC];
  sh[t] = caout[idx]; __syncthreads();
  const float* w = wo + (size_t)t * NC;
  float v = bo[t];
  for (int k = 0; k < NC; ++k) v += sh[k] * w[k];
  v += qbase[(size_t)q * NC + t];
  float mu  = block_sum256(v, red) * (1.0f / NC);
  float d   = v - mu;
  float var = block_sum256(d * d, red) * (1.0f / NC);
  float o = d * rsqrtf(var + 1e-5f) * g[t] + bln[t];
  x[idx] = o; xb[idx] = (bf16_t)o;
}

// FFN layer 1 (relu), WMMA K=256
__global__ void k_ffn1(const bf16_t* xb, const bf16_t* w1b, const float* b1, bf16_t* hbuf) {
  int w = blockIdx.x * 4 + (threadIdx.x >> 5);
  int ft = w % (NDFF / 16), qt = (w / (NDFF / 16)) % NQT, b = w / ((NDFF / 16) * NQT);
  const bf16_t* A = xb + ((size_t)b * NQP + qt * 16) * NC;
  v8f acc = {};
  for (int kk = 0; kk < NC / 32; ++kk) {
    v16bf a  = load_a_rm_bf16(A + kk * 32, NC);
    v16bf bm = load_b_cm_bf16(w1b + (size_t)(ft * 16) * NC + kk * 32, NC);
    acc = wmma_bf16(a, bm, acc);
  }
  int l = threadIdx.x & 31, n = l & 15, mh = l >> 4;
  float bias = b1[ft * 16 + n];
  bf16_t* dst = hbuf + ((size_t)b * NQP + qt * 16) * NDFF + ft * 16;
#pragma unroll
  for (int r = 0; r < 8; ++r) {
    int m = r + mh * 8;
    dst[(size_t)m * NDFF + n] = (bf16_t)fmaxf(acc[r] + bias, 0.f);
  }
}

// FFN layer 2, WMMA K=2048
__global__ void k_ffn2(const bf16_t* hbuf, const bf16_t* w2b, const float* b2, float* y) {
  int w = blockIdx.x * 4 + (threadIdx.x >> 5);
  int ct = w % (NC / 16), qt = (w / (NC / 16)) % NQT, b = w / ((NC / 16) * NQT);
  const bf16_t* A = hbuf + ((size_t)b * NQP + qt * 16) * NDFF;
  v8f acc = {};
  for (int kk = 0; kk < NDFF / 32; ++kk) {
    v16bf a  = load_a_rm_bf16(A + kk * 32, NDFF);
    v16bf bm = load_b_cm_bf16(w2b + (size_t)(ct * 16) * NDFF + kk * 32, NDFF);
    acc = wmma_bf16(a, bm, acc);
  }
  int l = threadIdx.x & 31, n = l & 15, mh = l >> 4;
  float bias = b2[ct * 16 + n];
  float* dst = y + ((size_t)b * NQP + qt * 16) * NC + ct * 16;
#pragma unroll
  for (int r = 0; r < 8; ++r) dst[(size_t)(r + mh * 8) * NC + n] = acc[r] + bias;
}

// residual + LN(n3) -> iou_tokens output; mask-key projection -> bf16
__global__ void k_ln3_mask(const float* x, const float* y, const float* g, const float* bln,
                           const float* mpw, const float* mpb, float* iou, bf16_t* mkb) {
  int bq = blockIdx.x, b = bq / NQP, q = bq % NQP, t = threadIdx.x;
  size_t idx = (size_t)bq * NC + t;
  if (q >= NQ) { mkb[idx] = (bf16_t)0.f; return; }
  __shared__ float sh[NC]; __shared__ float red[NC];
  float v = x[idx] + y[idx];
  float mu  = block_sum256(v, red) * (1.0f / NC);
  float d   = v - mu;
  float var = block_sum256(d * d, red) * (1.0f / NC);
  float tv  = d * rsqrtf(var + 1e-5f) * g[t] + bln[t];
  iou[((size_t)b * NQ + q) * NC + t] = tv;
  sh[t] = tv; __syncthreads();
  const float* w = mpw + (size_t)t * NC;
  float mk = mpb[t];
  for (int k = 0; k < NC; ++k) mk += sh[k] * w[k];
  mkb[idx] = (bf16_t)mk;
}

// mask logits: logits[b,q,n] = sum_c mk[b,q,c] * img[b,c,n]
__global__ void k_mask_logits(const bf16_t* mkb, const float* img, float* logits) {
  int w = blockIdx.x * 4 + (threadIdx.x >> 5);
  int nt = w % (NHW / 16), qt = (w / (NHW / 16)) % NQT, b = w / ((NHW / 16) * NQT);
  const bf16_t* A  = mkb + ((size_t)b * NQP + qt * 16) * NC;
  const float*  Bp = img + (size_t)b * NC * NHW + nt * 16;
  v8f acc = {};
  for (int kk = 0; kk < NC / 32; ++kk)
    acc = wmma_bf16(load_a_rm_bf16(A + kk * 32, NC),
                    load_b_rm_f32(Bp + (size_t)(kk * 32) * NHW, NHW), acc);
  int l = threadIdx.x & 31, n = l & 15, mh = l >> 4;
#pragma unroll
  for (int r = 0; r < 8; ++r) {
    int q = qt * 16 + r + mh * 8;
    if (q < NQ) logits[((size_t)b * NQ + q) * NHW + nt * 16 + n] = acc[r];
  }
}

// boxes from mask logits (sigmoid>0.5 <=> logit>0)
__global__ void k_boxes(const float* logits, float* boxes) {
  int bq = blockIdx.x, t = threadIdx.x;
  const float* L = logits + (size_t)bq * NHW;
  int xmin = NWW, xmax = -1, ymin = NHH, ymax = -1;
  for (int i = t; i < NHW; i += 256) {
    if (L[i] > 0.f) {
      int yy = i >> 6, xx = i & 63;
      xmin = min(xmin, xx); xmax = max(xmax, xx);
      ymin = min(ymin, yy); ymax = max(ymax, yy);
    }
  }
  __shared__ int rxm[256], rxM[256], rym[256], ryM[256];
  rxm[t] = xmin; rxM[t] = xmax; rym[t] = ymin; ryM[t] = ymax;
  __syncthreads();
  for (int s = 128; s > 0; s >>= 1) {
    if (t < s) {
      rxm[t] = min(rxm[t], rxm[t + s]); rxM[t] = max(rxM[t], rxM[t + s]);
      rym[t] = min(rym[t], rym[t + s]); ryM[t] = max(ryM[t], ryM[t + s]);
    }
    __syncthreads();
  }
  if (t == 0) {
    float* o = boxes + (size_t)bq * 4;
    if (rxM[0] < 0) { o[0] = 0.5f; o[1] = 0.5f; o[2] = 0.f; o[3] = 0.f; }
    else {
      float x0 = rxm[0] / 63.f, x1 = rxM[0] / 63.f;
      float y0 = rym[0] / 63.f, y1 = ryM[0] / 63.f;
      float cx = (x0 + x1) * 0.5f, cy = (y0 + y1) * 0.5f, ww = x1 - x0, hh = y1 - y0;
      o[0] = fminf(fmaxf(cx, 0.f), 1.f);
      o[1] = fminf(fmaxf(cy, 0.f), 1.f);
      o[2] = fminf(fmaxf(ww, 0.f), 1.f);
      o[3] = fminf(fmaxf(hh, 0.f), 1.f);
    }
  }
}

// ---------------- host launcher ----------------
extern "C" void kernel_launch(void* const* d_in, const int* in_sizes, int n_in,
                              void* d_out, int out_size, void* d_ws, size_t ws_size,
                              hipStream_t stream) {
  const float* img   = (const float*)d_in[0];
  const float* pe    = (const float*)d_in[1];
  const float* sa_wi = (const float*)d_in[2];
  const float* sa_bi = (const float*)d_in[3];
  const float* sa_wo = (const float*)d_in[4];
  const float* sa_bo = (const float*)d_in[5];
  const float* ca_wi = (const float*)d_in[6];
  const float* ca_bi = (const float*)d_in[7];
  const float* ca_wo = (const float*)d_in[8];
  const float* ca_bo = (const float*)d_in[9];
  const float* n1_g  = (const float*)d_in[10];
  const float* n1_b  = (const float*)d_in[11];
  const float* n2_g  = (const float*)d_in[12];
  const float* n2_b  = (const float*)d_in[13];
  const float* n3_g  = (const float*)d_in[14];
  const float* n3_b  = (const float*)d_in[15];
  const float* w1    = (const float*)d_in[16];
  const float* b1    = (const float*)d_in[17];
  const float* w2    = (const float*)d_in[18];
  const float* b2    = (const float*)d_in[19];
  const float* mpw   = (const float*)d_in[20];
  const float* mpb   = (const float*)d_in[21];
  (void)in_sizes; (void)n_in; (void)out_size; (void)ws_size;

  float* out    = (float*)d_out;
  float* iou    = out;                               // (B,Q,C)
  float* logits = iou + (size_t)NB * NQ * NC;        // (B,Q,H,W)
  float* boxes  = logits + (size_t)NB * NQ * NHW;    // (B,Q,4)

  char* cur = (char*)d_ws;
  auto alloc = [&](size_t bytes) -> void* {
    void* p = (void*)cur;
    cur += (bytes + 255) & ~(size_t)255;
    return p;
  };
  float*  sqkv  = (float*)alloc((size_t)3 * NQ * NC * 4);
  float*  attn1 = (float*)alloc((size_t)NQ * NC * 4);
  float*  qbase = (float*)alloc((size_t)NQ * NC * 4);
  bf16_t* qpca  = (bf16_t*)alloc((size_t)NQP * NC * 2);
  bf16_t* cawib = (bf16_t*)alloc((size_t)3 * NC * NC * 2);
  bf16_t* w1b   = (bf16_t*)alloc((size_t)NDFF * NC * 2);
  bf16_t* w2b   = (bf16_t*)alloc((size_t)NC * NDFF * 2);
  bf16_t* kpT   = (bf16_t*)alloc((size_t)NB * NC * NHW * 2);
  bf16_t* vpT   = (bf16_t*)alloc((size_t)NB * NC * NHW * 2);
  float*  caout = (float*)alloc((size_t)NB * NQP * NC * 4);
  float*  xbuf  = (float*)alloc((size_t)NB * NQP * NC * 4);
  bf16_t* xb    = (bf16_t*)alloc((size_t)NB * NQP * NC * 2);
  bf16_t* hbuf  = (bf16_t*)alloc((size_t)NB * NQP * NDFF * 2);
  float*  ybuf  = (float*)alloc((size_t)NB * NQP * NC * 4);
  bf16_t* mkb   = (bf16_t*)alloc((size_t)NB * NQP * NC * 2);

  // weight conversions
  k_f32_to_bf16<<<(3 * NC * NC + 255) / 256, 256, 0, stream>>>(ca_wi, cawib, 3 * NC * NC);
  k_f32_to_bf16<<<(NDFF * NC + 255) / 256, 256, 0, stream>>>(w1, w1b, NDFF * NC);
  k_f32_to_bf16<<<(NC * NDFF + 255) / 256, 256, 0, stream>>>(w2, w2b, NC * NDFF);

  // stage A (batch-independent)
  k_self_qkv<<<(3 * NQ * NC + 255) / 256, 256, 0, stream>>>(pe, sa_wi, sa_bi, sqkv);
  k_self_attn<<<NQ * NNH, 64, 0, stream>>>(sqkv, attn1);
  k_self_out_ln<<<NQ, NC, 0, stream>>>(attn1, sa_wo, sa_bo, pe, n2_g, n2_b, qbase);
  k_q_proj_ca<<<NQP, NC, 0, stream>>>(qbase, ca_wi, ca_bi, qpca);

  // stage B (per-batch, WMMA)
  k_kv_proj<<<(2 * NB * 16 * 256) / 4, 128, 0, stream>>>(img, cawib, ca_bi, kpT, vpT);
  k_cross_attn<<<NB * NNH * NQT, 32, 0, stream>>>(qpca, kpT, vpT, caout);
  k_oproj_ln<<<NB * NQP, NC, 0, stream>>>(caout, ca_wo, ca_bo, qbase, n1_g, n1_b, xbuf, xb);
  k_ffn1<<<(NB * NQT * (NDFF / 16)) / 4, 128, 0, stream>>>(xb, w1b, b1, hbuf);
  k_ffn2<<<(NB * NQT * (NC / 16)) / 4, 128, 0, stream>>>(hbuf, w2b, b2, ybuf);
  k_ln3_mask<<<NB * NQP, NC, 0, stream>>>(xbuf, ybuf, n3_g, n3_b, mpw, mpb, iou, mkb);
  k_mask_logits<<<(NB * NQT * (NHW / 16)) / 4, 128, 0, stream>>>(mkb, img, logits);
  k_boxes<<<NB * NQ, 256, 0, stream>>>(logits, boxes);
}
